// EGMN_24283745091917
// MI455X (gfx1250) — compile-verified, matrix operand order checked by reading, not run on hardware
//
#include <hip/hip_runtime.h>
#include <cstddef>

// ---------------------------------------------------------------------------
// Elman RNN (tanh): B=256, T=512, F=256, H=512 on gfx1250 (MI455X).
//
// Main path (needs ~130 MB workspace):
//   1) prep: pack [Wx; Wh] (768x512 f32) -> bf16 WMMA B-operand tiles (32x16).
//   2) xp-GEMM: xp[B*T, H] = x @ Wx (bf16, f32 accum) across the whole GPU.
//   3) scan: 16 independent workgroups (16 batch rows), 512 threads =
//      16 waves (4/SIMD -> guaranteed 256-VGPR budget; VRF is 1024/SIMD as
//      established by the round-6 allocation capping at v127 for 8 waves).
//      Wave w owns 32 output columns (2 slices): each LDS A tile feeds 2
//      WMMAs. Wh K-tiles 0..7 live in 128 resident VGPRs; K-tiles 8..15 are
//      streamed from L2 each step through a double buffer issued early
//      (opaque pointer prevents loop-invariant hoisting/spill). xp_t for
//      step t+1 is prefetched into per-lane registers and folded in at tanh.
// Fallback (small workspace): round-1 fused kernel.
//
// NOTE: EGMN_scan_xp is defined FIRST so the disassembly window shows its
// steady-state recurrence loop.
// ---------------------------------------------------------------------------

#define B_  256
#define T_  512
#define F_  256
#define H_  512
#define KTOT 768
#define NKT  24            // packed weight K-tiles (0..7 Wx, 8..23 Wh)
#define NKH  16            // Wh K-tiles
#define RES  8             // Wh K-tiles kept register-resident (per slice)
#define SDW  396           // fused-kernel A row stride in dwords (396%64=12)
#define AHDW 260           // scan h-buffer row stride in dwords (260%64=4)
#define XBDW 132           // xp-GEMM x-stage row stride in dwords (128+4)

typedef __attribute__((ext_vector_type(16))) __bf16 v16bf;
typedef __attribute__((ext_vector_type(8)))  float  v8f;

union Ubf {
    v16bf v;
    unsigned u[8];
    uint4 q[2];
};

__device__ __forceinline__ unsigned short f2bf(float f) {
    unsigned u = __float_as_uint(f);
    unsigned r = u + 0x7FFFu + ((u >> 16) & 1u);
    return (unsigned short)(r >> 16);
}
__device__ __forceinline__ unsigned packbf(float a, float b) {
    return (unsigned)f2bf(a) | ((unsigned)f2bf(b) << 16);
}
__device__ __forceinline__ float bf2f(unsigned lo16) {
    return __uint_as_float(lo16 << 16);
}
// Register-lean tanh: tanh(x) = sign(x) * (1 - 2/(e^{2|x|}+1)); exact at +-inf.
__device__ __forceinline__ float fast_tanh(float v) {
    float a = fabsf(v);
    float e = __expf(2.0f * a);
    float r = 1.0f - 2.0f * __builtin_amdgcn_rcpf(e + 1.0f);
    return __builtin_copysignf(r, v);
}
// CDNA5 16-bit A/B operand: dword d of the 8-VGPR group, lane half ->
// dword index inside a 32-wide K tile (V0..3: K0..7|K8..15, V4..7: K16..31).
__device__ __forceinline__ int kdw(int d, int half) {
    return d + ((d >= 4) ? 4 : 0) + 4 * half;
}

// ---------------------------------------------------------------------------
// Serial scan, xp precomputed. grid = 16, block = 512 (16 waves).
// Wave w owns columns [32w, 32w+32): 2 slices; Wh kt 0..7 resident (128
// VGPRs), kt 8..15 streamed from L2 per step (double-buffered, early issue).
// ---------------------------------------------------------------------------
__global__ __launch_bounds__(512)
void EGMN_scan_xp(const unsigned short* __restrict__ xp,
                  const float* __restrict__ bias,
                  const float* __restrict__ Wfc,
                  const float* __restrict__ bfc,
                  const unsigned* __restrict__ wws,
                  float* __restrict__ out) {
    __shared__ unsigned Ah[16 * AHDW];    // h as bf16 WMMA-A source (padded rows)
    __shared__ float    hf[16 * H_];      // final f32 h for Dense(1)

    const int tid   = threadIdx.x;
    const int w     = tid >> 5;           // 0..15 = column-tile pair
    const int lane  = tid & 31;
    const int half  = lane >> 4;
    const int mrow  = lane & 15;
    const int bRow0 = blockIdx.x * 16;
    const int n0    = w * 32 + mrow;      // slice 0 column; slice 1 = n0 + 16

    float bn[2];
    bn[0] = bias[n0];
    bn[1] = bias[n0 + 16];

    // Resident Wh: kt 0..RES-1 for both slices = 128 VGPRs.
    Ubf Bh[RES][2];
#pragma unroll
    for (int kt = 0; kt < RES; ++kt) {
#pragma unroll
        for (int s = 0; s < 2; ++s) {
            const uint4* p = (const uint4*)
                (wws + ((size_t)((8 + kt) * 32 + 2 * w + s)) * 256 + lane * 8);
            Bh[kt][s].q[0] = p[0];
            Bh[kt][s].q[1] = p[1];
        }
    }

    for (int i = tid; i < 16 * AHDW; i += 512) Ah[i] = 0u;   // h = 0

    // Per-lane xp gather base: element (s, r) lives at row (bRow0 + r + 8*half),
    // col (w*32 + s*16 + mrow); advance by H_ per t.
    const unsigned short* xbase =
        xp + (size_t)(bRow0 + 8 * half) * T_ * H_ + (size_t)n0;

    auto prefetch = [&](unsigned short (&xr)[2][8], int t) {
        const unsigned short* p = xbase + (size_t)t * H_;
#pragma unroll
        for (int s = 0; s < 2; ++s)
#pragma unroll
            for (int r = 0; r < 8; ++r)
                xr[s][r] = p[(size_t)r * T_ * H_ + s * 16];
    };

    auto step = [&](unsigned short (&xc)[2][8], unsigned short (&xn)[2][8], int t) {
        // Opaque base defeats loop-invariant hoisting of the streamed weights.
        const unsigned* wp = wws;
        asm volatile("" : "+s"(wp));
        const unsigned* pp = wp + (size_t)(2 * w) * 256 + lane * 8;

        auto loadStream = [&](Ubf (&d)[2], int kt) {
#pragma unroll
            for (int s = 0; s < 2; ++s) {
                const uint4* p = (const uint4*)(pp + (size_t)(8 + kt) * 8192 + s * 256);
                d[s].q[0] = p[0];
                d[s].q[1] = p[1];
            }
        };

        Ubf bs0[2], bs1[2];
        loadStream(bs0, RES);                  // first streamed kt, issued early
        if (t + 1 < T_) prefetch(xn, t + 1);   // hidden under this step's WMMAs
        __syncthreads();                       // previous h writes visible

        v8f acc[2] = {{}, {}};
        Ubf a;
        // Resident section: kt 0..RES-1 (streamed loads in flight meanwhile).
#pragma unroll
        for (int kt = 0; kt < RES; ++kt) {
            const int base = mrow * AHDW + kt * 16;
#pragma unroll
            for (int d = 0; d < 8; ++d) a.u[d] = Ah[base + kdw(d, half)];
            acc[0] = __builtin_amdgcn_wmma_f32_16x16x32_bf16(false, a.v, false, Bh[kt][0].v,
                                                             (short)0, acc[0], false, false);
            acc[1] = __builtin_amdgcn_wmma_f32_16x16x32_bf16(false, a.v, false, Bh[kt][1].v,
                                                             (short)0, acc[1], false, false);
        }
        // Streamed section: kt RES..15, double-buffered from L2.
#pragma unroll
        for (int j = 0; j < NKH - RES; ++j) {
            const int kt = RES + j;
            if (j + 1 < NKH - RES) {
                if (j & 1) loadStream(bs0, kt + 1);
                else       loadStream(bs1, kt + 1);
            }
            const int base = mrow * AHDW + kt * 16;
#pragma unroll
            for (int d = 0; d < 8; ++d) a.u[d] = Ah[base + kdw(d, half)];
            Ubf* bc = (j & 1) ? bs1 : bs0;
            acc[0] = __builtin_amdgcn_wmma_f32_16x16x32_bf16(false, a.v, false, bc[0].v,
                                                             (short)0, acc[0], false, false);
            acc[1] = __builtin_amdgcn_wmma_f32_16x16x32_bf16(false, a.v, false, bc[1].v,
                                                             (short)0, acc[1], false, false);
        }
        __syncthreads();                       // all reads of old h complete

        if (t < T_ - 1) {
            unsigned short* Ahs = (unsigned short*)Ah;
#pragma unroll
            for (int s = 0; s < 2; ++s)
#pragma unroll
                for (int r = 0; r < 8; ++r) {
                    const int m = r + half * 8;
                    const float hv =
                        fast_tanh(acc[s][r] + bn[s] + bf2f((unsigned)xc[s][r]));
                    Ahs[m * (AHDW * 2) + n0 + s * 16] = f2bf(hv);
                }
        } else {
#pragma unroll
            for (int s = 0; s < 2; ++s)
#pragma unroll
                for (int r = 0; r < 8; ++r) {
                    const int m = r + half * 8;
                    hf[m * H_ + n0 + s * 16] =
                        fast_tanh(acc[s][r] + bn[s] + bf2f((unsigned)xc[s][r]));
                }
        }
    };

    unsigned short xbA[2][8], xbB[2][8];
    prefetch(xbA, 0);
    for (int t = 0; t < T_; t += 2) {          // ping-pong: no register copies
        step(xbA, xbB, t);
        step(xbB, xbA, t + 1);
    }
    __syncthreads();

    // z = h_last @ Wfc + bfc - 0.05 ; wave w reduces row w (16 waves, 16 rows).
    {
        float s = 0.0f;
#pragma unroll 4
        for (int k = lane; k < H_; k += 32) s += hf[w * H_ + k] * Wfc[k];
#pragma unroll
        for (int off = 16; off > 0; off >>= 1) s += __shfl_xor(s, off, 32);
        if (lane == 0) out[bRow0 + w] = s + bfc[0] - 0.05f;
    }
}

// ---------------------------------------------------------------------------
// Pack W = [Wx; Wh] (768 x 512) into bf16 B-operand tiles, tile-major.
// ---------------------------------------------------------------------------
__global__ void EGMN_prep_weights(const float* __restrict__ Wx,
                                  const float* __restrict__ Wh,
                                  unsigned* __restrict__ wws) {
    const int tile = blockIdx.x;          // 0..767 (kt*32 + nt)
    const int kt = tile >> 5;
    const int nt = tile & 31;
    const int tid = threadIdx.x;          // 256
    const int lane = tid >> 3;
    const int j0 = (tid & 7) * 2;
    const int half = (lane >= 16) ? 8 : 0;
    const int n = nt * 16 + (lane & 15);
    const int k0 = kt * 32 + j0 + ((j0 >= 8) ? 8 : 0) + half;
    const int k1 = k0 + 1;
    const float v0 = (k0 < F_) ? Wx[k0 * H_ + n] : Wh[(k0 - F_) * H_ + n];
    const float v1 = (k1 < F_) ? Wx[k1 * H_ + n] : Wh[(k1 - F_) * H_ + n];
    wws[(size_t)tile * 256 + lane * 8 + (j0 >> 1)] = packbf(v0, v1);
}

// ---------------------------------------------------------------------------
// xp = x @ Wx (no bias; bias added in the scan). xp stored bf16 [B*T, H].
// grid = 2048 x 512 threads; block handles 64 flat rows, all 512 cols.
// ---------------------------------------------------------------------------
__global__ __launch_bounds__(512)
void EGMN_xp_gemm(const float* __restrict__ x,
                  const unsigned* __restrict__ wws,
                  unsigned short* __restrict__ xp) {
    __shared__ unsigned XB[64 * XBDW];

    const int tid  = threadIdx.x;
    const int w    = tid >> 5;            // 16 waves
    const int lane = tid & 31;
    const int half = lane >> 4;
    const int mrow = lane & 15;
    const size_t bt0 = (size_t)blockIdx.x * 64;

#pragma unroll
    for (int i = 0; i < 8; ++i) {
        const int idx = tid + 512 * i;
        const int m = idx >> 6;
        const int c = idx & 63;
        const float4 v = *(const float4*)(x + (bt0 + m) * F_ + c * 4);
        uint2 uu;
        uu.x = packbf(v.x, v.y);
        uu.y = packbf(v.z, v.w);
        *(uint2*)&XB[m * XBDW + c * 2] = uu;
    }
    __syncthreads();

    Ubf Bw[8][2];
#pragma unroll
    for (int kt = 0; kt < 8; ++kt) {
#pragma unroll
        for (int s = 0; s < 2; ++s) {
            const int nt = w + 16 * s;
            const uint4* p = (const uint4*)(wws + ((size_t)(kt * 32 + nt)) * 256 + lane * 8);
            Bw[kt][s].q[0] = p[0];
            Bw[kt][s].q[1] = p[1];
        }
    }

#pragma unroll
    for (int ms = 0; ms < 4; ++ms) {
        v8f acc0 = {}, acc1 = {};
#pragma unroll
        for (int kt = 0; kt < 8; ++kt) {
            Ubf a;
            const int base = (ms * 16 + mrow) * XBDW + kt * 16;
#pragma unroll
            for (int d = 0; d < 8; ++d) a.u[d] = XB[base + kdw(d, half)];
            acc0 = __builtin_amdgcn_wmma_f32_16x16x32_bf16(false, a.v, false, Bw[kt][0].v,
                                                           (short)0, acc0, false, false);
            acc1 = __builtin_amdgcn_wmma_f32_16x16x32_bf16(false, a.v, false, Bw[kt][1].v,
                                                           (short)0, acc1, false, false);
        }
#pragma unroll
        for (int r = 0; r < 8; ++r) {
            const int m = ms * 16 + r + half * 8;
            xp[(bt0 + m) * H_ + w * 16 + mrow]        = f2bf(acc0[r]);
            xp[(bt0 + m) * H_ + (w + 16) * 16 + mrow] = f2bf(acc1[r]);
        }
    }
}

// ---------------------------------------------------------------------------
// Fallback fused scan (round-1 design) for small workspaces.
// ---------------------------------------------------------------------------
__global__ __launch_bounds__(1024)
void EGMN_rnn_fused(const float* __restrict__ x,
                    const float* __restrict__ bias,
                    const float* __restrict__ Wfc,
                    const float* __restrict__ bfc,
                    const unsigned* __restrict__ wws,
                    float* __restrict__ out) {
    __shared__ unsigned Abuf[16 * SDW];
    __shared__ float    hf[16 * H_];

    const int tid   = threadIdx.x;
    const int w     = tid >> 5;
    const int lane  = tid & 31;
    const int half  = lane >> 4;
    const int mrow  = lane & 15;
    const int bRow0 = blockIdx.x * 16;
    const int nGlob = w * 16 + mrow;
    const float bn = bias[nGlob];

    Ubf Breg[NKT];
#pragma unroll
    for (int kt = 0; kt < NKT; ++kt) {
        const uint4* p = (const uint4*)(wws + ((size_t)(kt * 32 + w)) * 256 + lane * 8);
        Breg[kt].q[0] = p[0];
        Breg[kt].q[1] = p[1];
    }
    for (int i = tid; i < 16 * 256; i += 1024) {
        const int m = i >> 8;
        Abuf[m * SDW + 128 + (i & 255)] = 0u;
    }
    for (int t = 0; t < T_; ++t) {
        {
            const int m  = tid >> 6;
            const int f0 = (tid & 63) << 2;
            const float4 v = *(const float4*)(x + (((size_t)(bRow0 + m)) * T_ + t) * F_ + f0);
            uint2 uu;
            uu.x = packbf(v.x, v.y);
            uu.y = packbf(v.z, v.w);
            *(uint2*)&Abuf[m * SDW + (f0 >> 1)] = uu;
        }
        __syncthreads();
        v8f acc = {};
#pragma unroll
        for (int kt = 0; kt < NKT; ++kt) {
            Ubf a;
            const int base = mrow * SDW + kt * 16;
#pragma unroll
            for (int d = 0; d < 8; ++d) a.u[d] = Abuf[base + kdw(d, half)];
            acc = __builtin_amdgcn_wmma_f32_16x16x32_bf16(false, a.v, false, Breg[kt].v,
                                                          (short)0, acc, false, false);
        }
        __syncthreads();
        {
            unsigned short* Ahs = (unsigned short*)Abuf;
#pragma unroll
            for (int r = 0; r < 8; ++r) {
                const int m = r + half * 8;
                const float hv = fast_tanh(acc[r] + bn);
                Ahs[m * (SDW * 2) + F_ + nGlob] = f2bf(hv);
                if (t == T_ - 1) hf[m * H_ + nGlob] = hv;
            }
        }
    }
    __syncthreads();
    if (w < 16) {
        float s = 0.0f;
#pragma unroll 4
        for (int k = lane; k < H_; k += 32) s += hf[w * H_ + k] * Wfc[k];
#pragma unroll
        for (int off = 16; off > 0; off >>= 1) s += __shfl_xor(s, off, 32);
        if (lane == 0) out[bRow0 + w] = s + bfc[0] - 0.05f;
    }
}

// ---------------------------------------------------------------------------
extern "C" void kernel_launch(void* const* d_in, const int* in_sizes, int n_in,
                              void* d_out, int out_size, void* d_ws, size_t ws_size,
                              hipStream_t stream) {
    const float* x   = (const float*)d_in[0];
    const float* Wx  = (const float*)d_in[1];
    const float* Wh  = (const float*)d_in[2];
    const float* b   = (const float*)d_in[3];
    const float* Wfc = (const float*)d_in[4];
    const float* bfc = (const float*)d_in[5];
    float* out = (float*)d_out;

    unsigned* wws = (unsigned*)d_ws;                       // 768 KB weight tiles
    const size_t XPOFF = (size_t)1 << 20;                  // xp at +1 MB
    unsigned short* xpws = (unsigned short*)((char*)d_ws + XPOFF);
    const size_t need = XPOFF + (size_t)B_ * T_ * H_ * 2;  // ~129.8 MB

    EGMN_prep_weights<<<NKT * 32, 256, 0, stream>>>(Wx, Wh, wws);
    if (ws_size >= need) {
        EGMN_xp_gemm<<<(B_ * T_) / 64, 512, 0, stream>>>(x, wws, xpws);
        EGMN_scan_xp<<<B_ / 16, 512, 0, stream>>>(xpws, b, Wfc, bfc, wws, out);
    } else {
        EGMN_rnn_fused<<<B_ / 16, 1024, 0, stream>>>(x, b, Wfc, bfc, wws, out);
    }
}
